// StandaloneCustomMSDeformableAttention_2860448219636
// MI455X (gfx1250) — matrix operand order repaired and checked by reading, not compiled, not run on hardware
//
#include <hip/hip_runtime.h>
#include <cmath>

// ---------------- problem constants (from reference) ----------------
#define BS      4
#define NQ      10000
#define NV      13294            // 100*100 + 50*50 + 25*25 + 13*13
#define EMBED   256
#define HEADS   8
#define DH      32
#define LEVELS  4
#define POINTS  4
#define KDIM    256              // inner dim of every GEMM
#define MPROJ   384              // fused (Woff | Ww) output width

__constant__ __device__ int c_lvlW[LEVELS]  = {100, 50, 25, 13};
__constant__ __device__ int c_lvlH[LEVELS]  = {100, 50, 25, 13};
__constant__ __device__ int c_lvlS[LEVELS]  = {0, 10000, 12500, 13125};

typedef __attribute__((ext_vector_type(16))) __bf16 v16bf;
typedef __attribute__((ext_vector_type(8)))  float  v8f;
typedef __attribute__((ext_vector_type(4)))  float  f32x4;

// ---------------------------------------------------------------
// Weight pre-pass: f32 (K x M, row-major) -> bf16 transposed (M x K)
// so GEMM B-fragments become contiguous 32B loads per lane.
// ---------------------------------------------------------------
__global__ void wt_convert_kernel(const float* __restrict__ W,
                                  __bf16* __restrict__ Wt,
                                  int K, int M) {
    int idx = blockIdx.x * blockDim.x + threadIdx.x;
    if (idx >= K * M) return;
    int k = idx / M;
    int m = idx - k * M;
    Wt[(size_t)m * K + k] = (__bf16)W[(size_t)k * M + m];
}

// ---------------------------------------------------------------
// GEMM: C[nrows x Md] = A[nrows x 256] @ W[256 x Md] + bias (+ residual)
//   A   : f32, row-major;  Wt : bf16, pre-transposed (Md x 256)
// block = 128 threads (4 waves). Block tile = 64(M) x 64(N).
// Wave: 16 rows x 64 cols = 4 accumulators; k-loop step 32, manually
// double-buffered so next-tile loads overlap the 4 WMMAs.
// bias2: if non-null, bias for cols >= bias2_from (fused projections).
// ---------------------------------------------------------------
__global__ __launch_bounds__(128)
void gemm_bias_wmma_kernel(const float* __restrict__ A,
                           const __bf16* __restrict__ Wt,
                           const float* __restrict__ bias,
                           const float* __restrict__ bias2,
                           int bias2_from,
                           const float* __restrict__ residual,
                           float* __restrict__ C,
                           int nrows, int Md) {
    const int lane = threadIdx.x & 31;
    const int wv   = threadIdx.x >> 5;        // 0..3
    const int half = lane >> 4;               // 0 or 1
    const int r16  = lane & 15;

    const int rowBase = blockIdx.x * 64 + wv * 16;
    const int colBase = blockIdx.y * 64;
    const int aRow    = rowBase + r16;        // row this lane loads for A frag
    const bool aLive  = (aRow < nrows);
    const float* aBase = A + (size_t)aRow * KDIM + 16 * half;
    const __bf16* bBase = Wt + (size_t)(colBase + r16) * KDIM + 16 * half;

    v8f acc[4];
#pragma unroll
    for (int t = 0; t < 4; ++t)
#pragma unroll
        for (int i = 0; i < 8; ++i) acc[t][i] = 0.0f;

    // ---- fragment loaders ----
    // A layout (16-bit A, 16x32): half 0 -> K k..k+15,
    //                             half 1 -> K k+16..k+31
    auto loadA = [&](int k) -> v16bf {
        v16bf a;
        if (aLive) {
            const float* ap = aBase + k;
            f32x4 u0 = *(const f32x4*)(ap);
            f32x4 u1 = *(const f32x4*)(ap + 4);
            f32x4 u2 = *(const f32x4*)(ap + 8);
            f32x4 u3 = *(const f32x4*)(ap + 12);
#pragma unroll
            for (int i = 0; i < 4; ++i) {
                a[i]      = (__bf16)u0[i];
                a[4 + i]  = (__bf16)u1[i];
                a[8 + i]  = (__bf16)u2[i];
                a[12 + i] = (__bf16)u3[i];
            }
        } else {
#pragma unroll
            for (int i = 0; i < 16; ++i) a[i] = (__bf16)0.0f;
        }
        return a;
    };
    // B layout (16-bit B, 32x16): lane holds col (lane&15),
    // K = k + 16*half + 0..15, contiguous in pre-transposed weights.
    auto loadB = [&](int t, int k) -> v16bf {
        return *(const v16bf*)(bBase + (size_t)t * 16 * KDIM + k);
    };

    // ---- software-pipelined k loop ----
    v16bf aCur = loadA(0);
    v16bf bCur[4];
#pragma unroll
    for (int t = 0; t < 4; ++t) bCur[t] = loadB(t, 0);

#pragma unroll
    for (int k = 0; k < KDIM; k += 32) {
        v16bf aNxt = aCur;
        v16bf bNxt[4];
        if (k + 32 < KDIM) {
            aNxt = loadA(k + 32);
#pragma unroll
            for (int t = 0; t < 4; ++t) bNxt[t] = loadB(t, k + 32);
        } else {
#pragma unroll
            for (int t = 0; t < 4; ++t) bNxt[t] = bCur[t];
        }
#pragma unroll
        for (int t = 0; t < 4; ++t)
            acc[t] = __builtin_amdgcn_wmma_f32_16x16x32_bf16(
                false, aCur, false, bCur[t], (short)0, acc[t], false, false);
        aCur = aNxt;
#pragma unroll
        for (int t = 0; t < 4; ++t) bCur[t] = bNxt[t];
    }

    // ---- epilogue: D VGPR r holds (M = r + 8*half, N = lane&15) ----
#pragma unroll
    for (int t = 0; t < 4; ++t) {
        const int col = colBase + t * 16 + r16;
        const float bc = (bias2 && col >= bias2_from) ? bias2[col - bias2_from]
                                                      : bias[col];
#pragma unroll
        for (int r = 0; r < 8; ++r) {
            const int orow = rowBase + r + 8 * half;
            if (orow < nrows) {
                const size_t o = (size_t)orow * Md + col;
                float val = acc[t][r] + bc;
                if (residual) val += residual[(size_t)orow * KDIM + col];
                C[o] = val;
            }
        }
    }
}

// ---------------------------------------------------------------
// Deformable sampling + fused softmax.
// One wave per (b,q,h); lane = channel d in [0,32). Every bilinear
// corner fetch is one contiguous 128B load from the projected value
// tensor (L2-resident: 54MB < 192MB L2).
// projb: (BS,NQ,384) = [ offsets 256 | aw logits 128 ] per row.
// ---------------------------------------------------------------
__global__ __launch_bounds__(256)
void msda_sample_kernel(const float* __restrict__ v,      // (BS, NV, 256) f32
                        const float* __restrict__ projb,  // (BS, NQ, 384) f32
                        const float* __restrict__ refp,   // (BS, NQ, 4, 2) f32
                        float* __restrict__ msda) {       // (BS, NQ, 256) f32
    const int gid  = blockIdx.x * blockDim.x + threadIdx.x;
    const int wid  = gid >> 5;
    const int lane = threadIdx.x & 31;
    const int total = BS * NQ * HEADS;
    if (wid >= total) return;                 // whole-wave uniform

    const int h  = wid & (HEADS - 1);
    const int bq = wid >> 3;                  // b*NQ + q
    const int b  = bq / NQ;

    const float* row = projb + (size_t)bq * MPROJ;
    const float* op  = row + h * (LEVELS * POINTS * 2);      // 32 offsets
    const float* lg  = row + EMBED + h * (LEVELS * POINTS);  // 16 logits
    const float* rp  = refp + (size_t)bq * (LEVELS * 2);

    // fused softmax over the 16 (level,point) logits of this head
    float w[16];
    float mx = lg[0];
#pragma unroll
    for (int i = 0; i < 16; ++i) { w[i] = lg[i]; mx = fmaxf(mx, w[i]); }
    float s = 0.0f;
#pragma unroll
    for (int i = 0; i < 16; ++i) { w[i] = __expf(w[i] - mx); s += w[i]; }
    const float invs = 1.0f / s;

    float acc = 0.0f;
#pragma unroll
    for (int l = 0; l < LEVELS; ++l) {
        const int Wl = c_lvlW[l], Hl = c_lvlH[l], ls = c_lvlS[l];
        const float rx = rp[l * 2 + 0];
        const float ry = rp[l * 2 + 1];
        const float* vbase = v + ((size_t)(b * NV + ls) * HEADS + h) * DH;
#pragma unroll
        for (int p = 0; p < POINTS; ++p) {
            // loc = ref + off/(W,H);  x = loc.x*W - 0.5 = ref.x*W + off.x - 0.5
            const float x = rx * (float)Wl + op[(l * POINTS + p) * 2 + 0] - 0.5f;
            const float y = ry * (float)Hl + op[(l * POINTS + p) * 2 + 1] - 0.5f;
            const float x0f = floorf(x), y0f = floorf(y);
            const float fx = x - x0f, fy = y - y0f;
            const int x0 = (int)x0f, y0 = (int)y0f;
            const float aw = w[l * POINTS + p] * invs;
#pragma unroll
            for (int dy = 0; dy < 2; ++dy) {
                const int yi = y0 + dy;
                if (yi < 0 || yi >= Hl) continue;
                const float wy = dy ? fy : (1.0f - fy);
#pragma unroll
                for (int dx = 0; dx < 2; ++dx) {
                    const int xi = x0 + dx;
                    if (xi < 0 || xi >= Wl) continue;
                    const float wx = dx ? fx : (1.0f - fx);
                    const float* vp = vbase + (size_t)(yi * Wl + xi) * (HEADS * DH);
                    acc += aw * wy * wx * vp[lane];     // coalesced 128B / wave
                }
            }
        }
    }
    msda[(size_t)bq * EMBED + h * DH + lane] = acc;
}

// ---------------------------------------------------------------
// Host-side launcher
// ---------------------------------------------------------------
extern "C" void kernel_launch(void* const* d_in, const int* in_sizes, int n_in,
                              void* d_out, int out_size, void* d_ws, size_t ws_size,
                              hipStream_t stream) {
    (void)in_sizes; (void)n_in; (void)out_size; (void)ws_size;

    const float* query = (const float*)d_in[0];   // (BS,NQ,256)
    const float* value = (const float*)d_in[1];   // (BS,NV,256)
    const float* refp  = (const float*)d_in[2];   // (BS,NQ,4,2)
    // d_in[3] spatial_shapes (static, hardcoded)
    const float* Wv    = (const float*)d_in[4];
    const float* bv    = (const float*)d_in[5];
    const float* Woff  = (const float*)d_in[6];
    const float* boff  = (const float*)d_in[7];
    const float* Ww    = (const float*)d_in[8];
    const float* bw    = (const float*)d_in[9];
    const float* Wo    = (const float*)d_in[10];
    const float* bo    = (const float*)d_in[11];

    const int rowsV = BS * NV;   // 53176
    const int rowsQ = BS * NQ;   // 40000

    // workspace carve-up (all 256B aligned)
    char* ws = (char*)d_ws;
    size_t off = 0;
    auto carve = [&](size_t bytes) { char* p = ws + off; off += (bytes + 255) & ~(size_t)255; return p; };
    __bf16* WtV   = (__bf16*)carve(256 * 256 * sizeof(__bf16));
    __bf16* WtCat = (__bf16*)carve(MPROJ * 256 * sizeof(__bf16)); // [Woff^T ; Ww^T]
    __bf16* WtO   = (__bf16*)carve(256 * 256 * sizeof(__bf16));
    float*  vbuf  = (float*)carve((size_t)rowsV * 256 * sizeof(float));
    float*  projb = (float*)carve((size_t)rowsQ * MPROJ * sizeof(float));
    float*  msda  = (float*)carve((size_t)rowsQ * 256 * sizeof(float));

    // 1) weight transpose + bf16 convert (WtCat = concat along M: rows
    //    0..255 = Woff columns, rows 256..383 = Ww columns; layout is
    //    M x K row-major so concatenation is just adjacent regions)
    wt_convert_kernel<<<(256 * 256 + 255) / 256, 256, 0, stream>>>(Wv,   WtV,               256, 256);
    wt_convert_kernel<<<(256 * 256 + 255) / 256, 256, 0, stream>>>(Woff, WtCat,             256, 256);
    wt_convert_kernel<<<(256 * 128 + 255) / 256, 256, 0, stream>>>(Ww,   WtCat + 256 * 256, 256, 128);
    wt_convert_kernel<<<(256 * 256 + 255) / 256, 256, 0, stream>>>(Wo,   WtO,               256, 256);

    // 2) v = value @ Wv + bv
    gemm_bias_wmma_kernel<<<dim3((rowsV + 63) / 64, 256 / 64), 128, 0, stream>>>(
        value, WtV, bv, nullptr, 0, nullptr, vbuf, rowsV, 256);

    // 3) fused projections: projb = query @ [Woff|Ww] + [boff|bw]
    gemm_bias_wmma_kernel<<<dim3((rowsQ + 63) / 64, MPROJ / 64), 128, 0, stream>>>(
        query, WtCat, boff, bw, 256, nullptr, projb, rowsQ, MPROJ);

    // 4) deformable sampling (one wave per (b,q,head)), softmax fused
    const int totalWaves = BS * NQ * HEADS;                  // 320000
    msda_sample_kernel<<<(totalWaves * 32 + 255) / 256, 256, 0, stream>>>(
        vbuf, projb, refp, msda);

    // 5) out = msda @ Wo + bo + query   (residual fused)
    gemm_bias_wmma_kernel<<<dim3((rowsQ + 63) / 64, 256 / 64), 128, 0, stream>>>(
        msda, WtO, bo, nullptr, 0, query, (float*)d_out, rowsQ, 256);
}
